// VisionTransformer_20495583936794
// MI455X (gfx1250) — matrix operand order repaired and checked by reading, not compiled, not run on hardware
//
#include <hip/hip_runtime.h>
#include <hip/hip_bf16.h>
#include <math.h>

// ---------------------------------------------------------------------------
// ViT forward for gfx1250 (MI455X). All GEMMs run on v_wmma_f32_16x16x32_bf16
// (bf16 A/B converted in-register from fp32 via single v_perm_b32 per pair,
// fp32 accumulate). Compute-bound: ~1 TFLOP of matmul.
// ---------------------------------------------------------------------------

typedef __attribute__((ext_vector_type(16))) __bf16 v16bf;
typedef __attribute__((ext_vector_type(8)))  float  v8f;

union Frag { v16bf v; unsigned u[8]; };

// fp32x2 -> packed bf16x2 (truncation) in ONE v_perm_b32:
// dst = { hi[31:16], lo[31:16] }
__device__ __forceinline__ unsigned pack_bf16x2(float lo, float hi) {
  return __builtin_amdgcn_perm(__float_as_uint(hi), __float_as_uint(lo),
                               0x07060302u);
}

__device__ __forceinline__ v8f vzero8() {
  v8f z = {0.f, 0.f, 0.f, 0.f, 0.f, 0.f, 0.f, 0.f};
  return z;
}

__device__ __forceinline__ v8f wmma_bf16(const Frag& a, const Frag& b, v8f c) {
  return __builtin_amdgcn_wmma_f32_16x16x32_bf16(
      /*neg_a=*/false, a.v, /*neg_b=*/false, b.v,
      /*c_mod=*/(short)0, c, /*reuse_a=*/false, /*reuse_b=*/false);
}

// A fragment (16x32, lane = row M, ISA 7.12.2 16-bit A layout):
// lane = m + 16*half; VGPR v holds K pair at kbase = (v>>2)*16 + half*8 + (v&3)*2
__device__ __forceinline__ void load_a_frag(Frag& f, const float* base, int ld,
                                            int row0, int k0) {
  int lane = threadIdx.x & 31;
  int hf = lane >> 4;
  int r  = lane & 15;
  const float* p = base + (size_t)(row0 + r) * ld + k0 + hf * 8;
#pragma unroll
  for (int g = 0; g < 2; ++g) {
    const float* q = p + g * 16;
#pragma unroll
    for (int i = 0; i < 4; ++i)
      f.u[g * 4 + i] = pack_bf16x2(q[2 * i], q[2 * i + 1]);
  }
}

// B fragment (32x16, lane = column N), loaded from row-major W so B = W^T:
// lane = n + 16*half; VGPR v holds K pair at kbase = (v>>2)*16 + half*8 + (v&3)*2
__device__ __forceinline__ void load_b_frag(Frag& f, const float* Wrow, int ld,
                                            int n0, int k0) {
  int lane = threadIdx.x & 31;
  int hf = lane >> 4;
  int n  = lane & 15;
  const float* p = Wrow + (size_t)(n0 + n) * ld + k0 + hf * 8;
#pragma unroll
  for (int v = 0; v < 8; ++v)
    f.u[v] = pack_bf16x2(p[(v >> 2) * 16 + (v & 3) * 2], p[(v >> 2) * 16 + (v & 3) * 2 + 1]);
}

// ---------------------------------------------------------------------------
// Generic GEMM: C[M,Nout] = A[M,K] @ W[Nout,K]^T + bias (+GELU) (+residual)
// One wave computes a 32x64 macro-tile: 8 WMMAs per K-step of 32.
// ---------------------------------------------------------------------------
__global__ __launch_bounds__(32)
void gemm_k(const float* __restrict__ A, const float* __restrict__ W,
            const float* __restrict__ bias, const float* __restrict__ residual,
            float* __restrict__ C, int M, int Nout, int K, int gelu) {
  int nblk = blockIdx.x * 64;
  int mblk = blockIdx.y * 32;

  v8f acc[2][4];
#pragma unroll
  for (int i = 0; i < 2; ++i)
#pragma unroll
    for (int j = 0; j < 4; ++j) acc[i][j] = vzero8();

  for (int k0 = 0; k0 < K; k0 += 32) {
    Frag a[2], b[4];
    load_a_frag(a[0], A, K, mblk,      k0);
    load_a_frag(a[1], A, K, mblk + 16, k0);
#pragma unroll
    for (int j = 0; j < 4; ++j) load_b_frag(b[j], W, K, nblk + j * 16, k0);
#pragma unroll
    for (int i = 0; i < 2; ++i)
#pragma unroll
      for (int j = 0; j < 4; ++j) acc[i][j] = wmma_bf16(a[i], b[j], acc[i][j]);
    if (k0 + 32 < K) {
      __builtin_prefetch(A + (size_t)mblk * K + k0 + 32, 0, 1);
      __builtin_prefetch(W + (size_t)nblk * K + k0 + 32, 0, 1);
    }
  }

  int lane = threadIdx.x & 31;
  int hf = lane >> 4;
  int nl = lane & 15;
#pragma unroll
  for (int i = 0; i < 2; ++i)
#pragma unroll
    for (int j = 0; j < 4; ++j)
#pragma unroll
      for (int r = 0; r < 8; ++r) {
        int m   = mblk + i * 16 + hf * 8 + r;
        int col = nblk + j * 16 + nl;
        float vv = acc[i][j][r] + bias[col];
        if (gelu) vv = 0.5f * vv * (1.0f + erff(vv * 0.70710678118654752f));
        if (residual) vv += residual[(size_t)m * Nout + col];
        C[(size_t)m * Nout + col] = vv;
      }
}

// ---------------------------------------------------------------------------
// Patch embedding: tok = unfold(x) @ patch_w^T + patch_b + pos_encoding
// A gathered from x (pairs along pj are contiguous); M=4096, K=256, N=1024.
// ---------------------------------------------------------------------------
__device__ __forceinline__ void load_patch_frag(Frag& f, const float* x,
                                                int row0, int k0) {
  int lane = threadIdx.x & 31;
  int hf = lane >> 4;
  int r  = lane & 15;
  int m  = row0 + r;
  int bi = m >> 10;
  int n  = m & 1023;
  int hp = n >> 5, wp = n & 31;
  const float* base = x + ((size_t)bi * 512 + (size_t)hp * 16) * 512 + wp * 16;
#pragma unroll
  for (int g = 0; g < 2; ++g)
#pragma unroll
    for (int i = 0; i < 4; ++i) {
      int k = k0 + g * 16 + hf * 8 + i * 2;           // k = pi*16 + pj
      const float* q = base + (k >> 4) * 512 + (k & 15);
      f.u[g * 4 + i] = pack_bf16x2(q[0], q[1]);
    }
}

__global__ __launch_bounds__(32)
void patch_embed_k(const float* __restrict__ x, const float* __restrict__ pw,
                   const float* __restrict__ pb, float* __restrict__ h) {
  int nblk = blockIdx.x * 64;
  int mblk = blockIdx.y * 32;

  v8f acc[2][4];
#pragma unroll
  for (int i = 0; i < 2; ++i)
#pragma unroll
    for (int j = 0; j < 4; ++j) acc[i][j] = vzero8();

  for (int k0 = 0; k0 < 256; k0 += 32) {
    Frag a[2], b[4];
    load_patch_frag(a[0], x, mblk,      k0);
    load_patch_frag(a[1], x, mblk + 16, k0);
#pragma unroll
    for (int j = 0; j < 4; ++j) load_b_frag(b[j], pw, 256, nblk + j * 16, k0);
#pragma unroll
    for (int i = 0; i < 2; ++i)
#pragma unroll
      for (int j = 0; j < 4; ++j) acc[i][j] = wmma_bf16(a[i], b[j], acc[i][j]);
  }

  int lane = threadIdx.x & 31;
  int hf = lane >> 4;
  int nl = lane & 15;
#pragma unroll
  for (int i = 0; i < 2; ++i)
#pragma unroll
    for (int j = 0; j < 4; ++j)
#pragma unroll
      for (int r = 0; r < 8; ++r) {
        int m    = mblk + i * 16 + hf * 8 + r;
        int col  = nblk + j * 16 + nl;
        int tokn = m & 1023;
        // pe[n,d]: freq = 10000^((d & ~1)/512), even->sin, odd->cos
        float freq = __powf(10000.0f, (float)(col & ~1) * (1.0f / 512.0f));
        float arg  = (float)tokn / freq;
        float pe   = (col & 1) ? __cosf(arg) : __sinf(arg);
        h[(size_t)m * 1024 + col] = acc[i][j][r] + pb[col] + pe;
      }
}

// ---------------------------------------------------------------------------
// LayerNorm over last dim (D=1024), one 256-thread block per row.
// ---------------------------------------------------------------------------
__global__ void layernorm_k(const float* __restrict__ x, const float* __restrict__ w,
                            const float* __restrict__ bvec, float* __restrict__ y) {
  const int D = 1024;
  int row = blockIdx.x;
  const float* xr = x + (size_t)row * D;
  float s = 0.f, s2 = 0.f;
  for (int i = threadIdx.x; i < D; i += 256) {
    float v = xr[i];
    s += v; s2 += v * v;
  }
#pragma unroll
  for (int m = 16; m >= 1; m >>= 1) {
    s  += __shfl_xor(s,  m, 32);
    s2 += __shfl_xor(s2, m, 32);
  }
  __shared__ float rs[8], rs2[8];
  int wid = threadIdx.x >> 5;
  if ((threadIdx.x & 31) == 0) { rs[wid] = s; rs2[wid] = s2; }
  __syncthreads();
  s = 0.f; s2 = 0.f;
#pragma unroll
  for (int i = 0; i < 8; ++i) { s += rs[i]; s2 += rs2[i]; }
  float mu  = s * (1.0f / 1024.0f);
  float var = s2 * (1.0f / 1024.0f) - mu * mu;
  float inv = rsqrtf(var + 1e-5f);
  for (int i = threadIdx.x; i < D; i += 256)
    y[(size_t)row * D + i] = (xr[i] - mu) * inv * w[i] + bvec[i];
}

// ---------------------------------------------------------------------------
// Flash attention: one wave per (batch, head, 16-row Q tile). 32 keys/step:
// S (16x32) via 8 WMMAs, online softmax, P (16x32) routed through LDS into
// A-layout, then P@V into 8 accumulator tiles (8 WMMAs). hd = 128.
// ---------------------------------------------------------------------------
__global__ __launch_bounds__(32)
void attention_k(const float* __restrict__ qkv, float* __restrict__ o) {
  const int N = 1024, D = 1024, LDQ = 3072, HD = 128;
  const float scale = 0.08838834764831845f;  // 1/sqrt(128)
  int qt   = blockIdx.x;   // 0..63
  int head = blockIdx.y;   // 0..7
  int bi   = blockIdx.z;   // 0..3
  int lane = threadIdx.x & 31;
  int hf = lane >> 4;
  int nl = lane & 15;

  const float* qbase = qkv + ((size_t)bi * N) * LDQ + head * HD;
  const float* kbase = qbase + D;
  const float* vbase = qbase + 2 * D;
  int q0 = qt * 16;

  Frag qf[4];
#pragma unroll
  for (int c = 0; c < 4; ++c) load_a_frag(qf[c], qbase, LDQ, q0, c * 32);

  v8f oacc[8];
  float mst[8], lst[8];
#pragma unroll
  for (int j = 0; j < 8; ++j) oacc[j] = vzero8();
#pragma unroll
  for (int r = 0; r < 8; ++r) { mst[r] = -1e30f; lst[r] = 0.f; }

  __shared__ float pshm[16 * 32];

  for (int kt = 0; kt < N; kt += 32) {
    v8f s0 = vzero8(), s1 = vzero8();
    {
      Frag kf;
#pragma unroll
      for (int c = 0; c < 4; ++c) {
        load_b_frag(kf, kbase, LDQ, kt, c * 32);
        s0 = wmma_bf16(qf[c], kf, s0);
      }
#pragma unroll
      for (int c = 0; c < 4; ++c) {
        load_b_frag(kf, kbase, LDQ, kt + 16, c * 32);
        s1 = wmma_bf16(qf[c], kf, s1);
      }
    }

    // online softmax over the 32 keys; row r lives across the 16 lanes of a half
    float p0[8], p1[8];
#pragma unroll
    for (int r = 0; r < 8; ++r) {
      float a = s0[r] * scale, b = s1[r] * scale;
      float mx = fmaxf(a, b);
#pragma unroll
      for (int m = 8; m >= 1; m >>= 1) mx = fmaxf(mx, __shfl_xor(mx, m, 32));
      float mn    = fmaxf(mst[r], mx);
      float alpha = __expf(mst[r] - mn);
      mst[r] = mn;
      float e0 = __expf(a - mn), e1 = __expf(b - mn);
      float rsum = e0 + e1;
#pragma unroll
      for (int m = 8; m >= 1; m >>= 1) rsum += __shfl_xor(rsum, m, 32);
      lst[r] = lst[r] * alpha + rsum;
      p0[r] = e0; p1[r] = e1;
#pragma unroll
      for (int j = 0; j < 8; ++j) oacc[j][r] *= alpha;
    }

    // C-layout -> LDS -> A-layout for P (16 rows x 32 keys)
    __syncthreads();
#pragma unroll
    for (int r = 0; r < 8; ++r) {
      int row = hf * 8 + r;
      pshm[row * 32 + nl]      = p0[r];
      pshm[row * 32 + 16 + nl] = p1[r];
    }
    __syncthreads();

    Frag pf;
#pragma unroll
    for (int g = 0; g < 2; ++g)
#pragma unroll
      for (int i = 0; i < 4; ++i) {
        int k = g * 16 + hf * 8 + i * 2;
        pf.u[g * 4 + i] = pack_bf16x2(pshm[nl * 32 + k], pshm[nl * 32 + k + 1]);
      }

#pragma unroll
    for (int j = 0; j < 8; ++j) {
      Frag vf;
#pragma unroll
      for (int v = 0; v < 8; ++v) {
        int k = (v >> 2) * 16 + hf * 8 + (v & 3) * 2;
        const float* vp = vbase + (size_t)(kt + k) * LDQ + j * 16 + nl;
        vf.u[v] = pack_bf16x2(vp[0], vp[LDQ]);
      }
      oacc[j] = wmma_bf16(pf, vf, oacc[j]);
    }
  }

#pragma unroll
  for (int j = 0; j < 8; ++j)
#pragma unroll
    for (int r = 0; r < 8; ++r) {
      int row = q0 + hf * 8 + r;
      int col = head * HD + j * 16 + nl;
      o[((size_t)bi * N + row) * D + col] = oacc[j][r] / lst[r];
    }
}

// ---------------------------------------------------------------------------
// Pixel-unshuffle reassembly: out[bi,ch,hp*16+i,wp*16+j] = h[bi, hp*32+wp, (i*16+j)*4+ch]
// ---------------------------------------------------------------------------
__global__ void unshuffle_k(const float* __restrict__ h, float* __restrict__ out) {
  int idx = blockIdx.x * 256 + threadIdx.x;
  if (idx >= 4 * 4 * 512 * 512) return;
  int ww = idx & 511;
  int t  = idx >> 9;
  int hh = t & 511;  t >>= 9;
  int ch = t & 3;
  int bi = t >> 2;
  int hp = hh >> 4, ii = hh & 15, wp = ww >> 4, jj = ww & 15;
  int n = hp * 32 + wp;
  int d = (ii * 16 + jj) * 4 + ch;
  out[idx] = h[((size_t)bi * 1024 + n) * 1024 + d];
}

// ---------------------------------------------------------------------------
extern "C" void kernel_launch(void* const* d_in, const int* in_sizes, int n_in,
                              void* d_out, int out_size, void* d_ws, size_t ws_size,
                              hipStream_t stream) {
  const float* x       = (const float*)d_in[0];
  const float* patch_w = (const float*)d_in[1];
  const float* patch_b = (const float*)d_in[2];
  const float* ln1_w   = (const float*)d_in[3];
  const float* ln1_b   = (const float*)d_in[4];
  const float* qkv_w   = (const float*)d_in[5];
  const float* qkv_b   = (const float*)d_in[6];
  const float* proj_w  = (const float*)d_in[7];
  const float* proj_b  = (const float*)d_in[8];
  const float* ln2_w   = (const float*)d_in[9];
  const float* ln2_b   = (const float*)d_in[10];
  const float* fc1_w   = (const float*)d_in[11];
  const float* fc1_b   = (const float*)d_in[12];
  const float* fc2_w   = (const float*)d_in[13];
  const float* fc2_b   = (const float*)d_in[14];

  // workspace layout (floats): 128 MB total
  float* h   = (float*)d_ws;               //  4M  (b*N*D)
  float* y   = h   + (4u  << 20);          //  4M  LN output
  float* qkv = y   + (4u  << 20);          // 12M  (b*N*3D)
  float* o   = qkv + (12u << 20);          //  4M  attention out
  float* f1  = o   + (4u  << 20);          //  8M  fc1/GELU out

  const int D = 1024;
  dim3 wv(32);

  patch_embed_k<<<dim3(16, 128), wv, 0, stream>>>(x, patch_w, patch_b, h);

  for (int l = 0; l < 12; ++l) {
    layernorm_k<<<4096, 256, 0, stream>>>(h, ln1_w + l * D, ln1_b + l * D, y);
    gemm_k<<<dim3(48, 128), wv, 0, stream>>>(
        y, qkv_w + (size_t)l * 3 * D * D, qkv_b + (size_t)l * 3 * D,
        nullptr, qkv, 4096, 3 * D, D, 0);
    attention_k<<<dim3(64, 8, 4), wv, 0, stream>>>(qkv, o);
    gemm_k<<<dim3(16, 128), wv, 0, stream>>>(
        o, proj_w + (size_t)l * D * D, proj_b + (size_t)l * D,
        h, h, 4096, D, D, 0);
    layernorm_k<<<4096, 256, 0, stream>>>(h, ln2_w + l * D, ln2_b + l * D, y);
    gemm_k<<<dim3(32, 128), wv, 0, stream>>>(
        y, fc1_w + (size_t)l * 2 * D * D, fc1_b + (size_t)l * 2 * D,
        nullptr, f1, 4096, 2 * D, D, 1);
    gemm_k<<<dim3(16, 128), wv, 0, stream>>>(
        f1, fc2_w + (size_t)l * D * 2 * D, fc2_b + (size_t)l * D,
        h, h, 4096, D, 2 * D, 0);
  }

  unshuffle_k<<<(4 * 4 * 512 * 512 + 255) / 256, 256, 0, stream>>>(h, (float*)d_out);
}
